// LSTMModel_70944269795690
// MI455X (gfx1250) — matrix operand order repaired and checked by reading, not compiled, not run on hardware
//
#include <hip/hip_runtime.h>
#include <stdint.h>

#define B_BATCH 4096
#define T_SEQ   50

typedef __attribute__((ext_vector_type(16))) __bf16 bf16x16;
typedef __attribute__((ext_vector_type(8)))  float  f32x8;

union Frag { bf16x16 f; uint4 u[2]; };

// Branch-free activations on native v_exp_f32 / v_rcp_f32 (no libm EXEC games).
__device__ __forceinline__ float sigf(float x)
{
    return __builtin_amdgcn_rcpf(1.0f + __expf(-x));
}
__device__ __forceinline__ float tanh_fast(float x)
{
    // tanh(x) = 1 - 2/(exp(2x)+1);  exp->0 gives -1, exp->inf gives +1.
    float e = __expf(2.0f * x);
    return 1.0f - 2.0f * __builtin_amdgcn_rcpf(e + 1.0f);
}

// ---------------------------------------------------------------------------
// Pack fp32 weight [K,N] (row-major) into WMMA bf16 B-operand fragments.
// Fragment (kt,nt): 32 lanes x 16 bf16.  Lane l: col n = l%16, g = l/16,
// element e holds W[kt*32 + g*8 + (e&7) + 16*(e>>3)][nt*16 + n]  (ISA 7.12.2/7.12.5 B layout).
// Rows >= srcK are zero (K padding, used for embed dim 50 -> 64).
// ---------------------------------------------------------------------------
__global__ void pack_w(const float* __restrict__ src, __bf16* __restrict__ dst,
                       int srcK, int K, int N)
{
    int tid   = blockIdx.x * blockDim.x + threadIdx.x;
    int total = (K >> 5) * (N >> 4) * 512;
    if (tid >= total) return;
    int frag = tid >> 9;
    int r    = tid & 511;
    int lane = r >> 4;
    int e    = r & 15;
    int NT   = N >> 4;
    int kt = frag / NT, nt = frag - kt * NT;
    int n = lane & 15, g = lane >> 4;
    int k   = kt * 32 + g * 8 + (e & 7) + 16 * (e >> 3);
    int col = nt * 16 + n;
    float v = (k < srcK) ? src[k * N + col] : 0.0f;
    dst[tid] = (__bf16)v;
}

// ---------------------------------------------------------------------------
// Embedding gather, fp32 -> bf16, pad feature dim 50 -> 64 with zeros.
// ---------------------------------------------------------------------------
__global__ void embed_gather(const int* __restrict__ idx, const float* __restrict__ emb,
                             __bf16* __restrict__ xe, int rows)
{
    int tid = blockIdx.x * blockDim.x + threadIdx.x;
    int row = tid >> 6;
    if (row >= rows) return;
    int col = tid & 63;
    int id  = idx[row];
    float v = (col < 50) ? emb[id * 50 + col] : 0.0f;
    xe[(size_t)row * 64 + col] = (__bf16)v;
}

// ---------------------------------------------------------------------------
// Fused LSTM layer.  Combined weights [Wi;Wh] (packed fragments) are staged
// once into LDS per block.  Each wave owns a 16-row batch tile and a private
// LDS A-buffer holding [x_t , h] concatenated; per step a unified K loop does
// g = [x_t,h] @ [Wi;Wh] + b with 4 independent gate WMMA chains.  c stays in
// VGPRs (ht loop fully unrolled, static indices); kt/t loops are unroll-1 to
// keep register pressure low (no spills).
// ---------------------------------------------------------------------------
template<int DIN, int H, int WAVES>
__global__ __launch_bounds__(WAVES * 32) void lstm_layer(
    const __bf16* __restrict__ xin,   // [B, T, DIN] bf16
    const __bf16* __restrict__ Wc,    // packed ((DIN+H)/32) x (4H/16) fragments
    const float*  __restrict__ bias,  // [4H] fp32
    __bf16* __restrict__ hout,        // [B,T,H] (storeAll) or [B,H] (last only)
    int T, int storeAll)
{
    constexpr int KTX = DIN / 32, KTH = H / 32, KTOT = KTX + KTH;
    constexpr int HT = H / 16, NT = 4 * HT;
    constexpr int AP = DIN + H + 8;          // A-buffer row stride (16B aligned)
    extern __shared__ __align__(16) char smem_raw[];
    __bf16* wlds  = (__bf16*)smem_raw;                    // KTOT*NT*512 elems
    __bf16* abase = wlds + (size_t)KTOT * NT * 512;       // WAVES*16*AP elems

    const int wave = threadIdx.x >> 5;
    const int lane = threadIdx.x & 31;
    const int n = lane & 15, g = lane >> 4;
    const int rowBase = (blockIdx.x * WAVES + wave) * 16;
    __bf16* abuf = abase + (size_t)wave * 16 * AP;

    // Block-cooperative copy: all layer weights global -> LDS (L2 -> LDS once).
    {
        const uint4* src = (const uint4*)Wc;
        uint4* dst = (uint4*)wlds;
        const int chunks = KTOT * NT * 64;   // 16B chunks
        for (int i = threadIdx.x; i < chunks; i += WAVES * 32) dst[i] = src[i];
    }
    // Zero A-buffer (h columns must start at 0).
    for (int i = lane; i < 16 * AP; i += 32) abuf[i] = (__bf16)0.0f;
    __syncthreads();

    float c[HT][8];
    #pragma unroll
    for (int ht = 0; ht < HT; ++ht)
        #pragma unroll
        for (int r = 0; r < 8; ++r) c[ht][r] = 0.0f;

    // Per-lane A-operand base (dense A layout: lo half K=kt*32+8g+e, hi +16).
    const __bf16* arow = abuf + n * AP + g * 8;

    #pragma unroll 1
    for (int t = 0; t < T; ++t) {
        // Stage x_t (16 rows x DIN) into A-buffer cols [0, DIN).
        constexpr int XCH = DIN / 8;         // 16B chunks per row
        for (int i = lane; i < 16 * XCH; i += 32) {
            int row = i / XCH, ck = i - row * XCH;
            *(uint4*)(abuf + row * AP + ck * 8) =
                *(const uint4*)(xin + ((size_t)(rowBase + row) * T + t) * DIN + ck * 8);
        }

        #pragma unroll
        for (int ht = 0; ht < HT; ++ht) {
            float bi = bias[0 * H + ht * 16 + n];
            float bf = bias[1 * H + ht * 16 + n];
            float bg = bias[2 * H + ht * 16 + n];
            float bo = bias[3 * H + ht * 16 + n];
            f32x8 a0, a1, a2, a3;
            #pragma unroll
            for (int r = 0; r < 8; ++r) { a0[r] = bi; a1[r] = bf; a2[r] = bg; a3[r] = bo; }

            #pragma unroll 1
            for (int kt = 0; kt < KTOT; ++kt) {
                Frag af;
                af.u[0] = *(const uint4*)(arow + kt * 32);        // ds_load_b128
                af.u[1] = *(const uint4*)(arow + kt * 32 + 16);
                const __bf16* wp = wlds + (size_t)(kt * NT + ht) * 512 + lane * 16;
                Frag w0, w1, w2, w3;
                w0.u[0] = *(const uint4*)(wp);
                w0.u[1] = *(const uint4*)(wp + 8);
                w1.u[0] = *(const uint4*)(wp + (size_t)HT * 512);
                w1.u[1] = *(const uint4*)(wp + (size_t)HT * 512 + 8);
                w2.u[0] = *(const uint4*)(wp + (size_t)2 * HT * 512);
                w2.u[1] = *(const uint4*)(wp + (size_t)2 * HT * 512 + 8);
                w3.u[0] = *(const uint4*)(wp + (size_t)3 * HT * 512);
                w3.u[1] = *(const uint4*)(wp + (size_t)3 * HT * 512 + 8);
                a0 = __builtin_amdgcn_wmma_f32_16x16x32_bf16(false, af.f, false, w0.f, (short)0, a0, false, false);
                a1 = __builtin_amdgcn_wmma_f32_16x16x32_bf16(false, af.f, false, w1.f, (short)0, a1, false, false);
                a2 = __builtin_amdgcn_wmma_f32_16x16x32_bf16(false, af.f, false, w2.f, (short)0, a2, false, false);
                a3 = __builtin_amdgcn_wmma_f32_16x16x32_bf16(false, af.f, false, w3.f, (short)0, a3, false, false);
            }

            // Gate nonlinearities; acc element r -> row (r + 8g), col (ht*16 + n).
            #pragma unroll
            for (int r = 0; r < 8; ++r) {
                float ig = sigf(a0[r]);
                float fg = sigf(a1[r]);
                float gg = tanh_fast(a2[r]);
                float og = sigf(a3[r]);
                float cn = fg * c[ht][r] + ig * gg;
                c[ht][r] = cn;
                float hv = og * tanh_fast(cn);
                abuf[(r + 8 * g) * AP + DIN + ht * 16 + n] = (__bf16)hv;   // ds_store_b16
            }
        }

        if (storeAll || t == T - 1) {        // uniform branch: EXEC stays all-ones
            constexpr int CH = H / 8;        // 16B chunks per row
            for (int i = lane; i < 16 * CH; i += 32) {
                int rr = i / CH, ck = i - rr * CH;
                uint4 v = *(const uint4*)(abuf + rr * AP + DIN + ck * 8);
                size_t off = storeAll ? (((size_t)(rowBase + rr) * T + t) * H + ck * 8)
                                      : ((size_t)(rowBase + rr) * H + ck * 8);
                *(uint4*)(hout + off) = v;
            }
        }
    }
}

// ---------------------------------------------------------------------------
// Dense + ReLU:  out[M,N] = relu(A[M,K] * W + b), bf16 in/out, fp32 acc.
// One wave per 16-row tile; A fragments hoisted across the N loop, N loop is
// unroll-1 so only 4 weight fragments + one accumulator are live.
// ---------------------------------------------------------------------------
template<int K, int N, int WAVES>
__global__ __launch_bounds__(WAVES * 32) void dense_relu(
    const __bf16* __restrict__ A, const __bf16* __restrict__ Wp,
    const float* __restrict__ bias, __bf16* __restrict__ out, int M)
{
    constexpr int KT = K / 32, NT = N / 16;
    const int wave = threadIdx.x >> 5;
    const int lane = threadIdx.x & 31;
    const int n = lane & 15, g = lane >> 4;
    const int rowBase = (blockIdx.x * WAVES + wave) * 16;
    if (rowBase >= M) return;

    __builtin_prefetch(Wp, 0, 3);            // global_prefetch: weights stay in L2

    Frag a[KT];
    const __bf16* ap = A + (size_t)(rowBase + n) * K + g * 8;
    #pragma unroll
    for (int kt = 0; kt < KT; ++kt) {
        a[kt].u[0] = *(const uint4*)(ap + kt * 32);
        a[kt].u[1] = *(const uint4*)(ap + kt * 32 + 16);
    }
    #pragma unroll 1
    for (int nt = 0; nt < NT; ++nt) {
        f32x8 acc;
        float bv = bias[nt * 16 + n];
        #pragma unroll
        for (int r = 0; r < 8; ++r) acc[r] = bv;
        #pragma unroll
        for (int kt = 0; kt < KT; ++kt) {
            Frag wf;
            const __bf16* wp = Wp + ((size_t)(kt * NT + nt) * 512 + lane * 16);
            wf.u[0] = *(const uint4*)(wp);
            wf.u[1] = *(const uint4*)(wp + 8);
            acc = __builtin_amdgcn_wmma_f32_16x16x32_bf16(
                false, a[kt].f, false, wf.f, (short)0, acc, false, false);
        }
        #pragma unroll
        for (int r = 0; r < 8; ++r) {
            float v = fmaxf(acc[r], 0.0f);
            out[(size_t)(rowBase + r + 8 * g) * N + nt * 16 + n] = (__bf16)v;
        }
    }
}

// ---------------------------------------------------------------------------
// Head: relu(h3_last @ Wd3 + bd3) @ Wd4 + bd4 -> log_softmax(2).
// ---------------------------------------------------------------------------
__global__ void head_kernel(const __bf16* __restrict__ h3, const float* __restrict__ Wd3,
                            const float* __restrict__ bd3, const float* __restrict__ Wd4,
                            const float* __restrict__ bd4, float* __restrict__ out, int B)
{
    int b = blockIdx.x * blockDim.x + threadIdx.x;
    if (b >= B) return;
    float v[32];
    #pragma unroll
    for (int k = 0; k < 32; ++k) v[k] = (float)h3[b * 32 + k];
    float l0 = bd4[0], l1 = bd4[1];
    for (int j = 0; j < 64; ++j) {
        float s = bd3[j];
        #pragma unroll
        for (int k = 0; k < 32; ++k) s += v[k] * Wd3[k * 64 + j];
        s = fmaxf(s, 0.0f);
        l0 += s * Wd4[j * 2 + 0];
        l1 += s * Wd4[j * 2 + 1];
    }
    float m   = fmaxf(l0, l1);
    float lse = m + __logf(__expf(l0 - m) + __expf(l1 - m));
    out[b * 2 + 0] = l0 - lse;
    out[b * 2 + 1] = l1 - lse;
}

// ---------------------------------------------------------------------------
extern "C" void kernel_launch(void* const* d_in, const int* in_sizes, int n_in,
                              void* d_out, int out_size, void* d_ws, size_t ws_size,
                              hipStream_t stream)
{
    (void)in_sizes; (void)n_in; (void)out_size; (void)ws_size;

    const int*   x_batch = (const int*)  d_in[0];
    const float* embed   = (const float*)d_in[1];
    const float* Wi1 = (const float*)d_in[2];
    const float* Wh1 = (const float*)d_in[3];
    const float* b1  = (const float*)d_in[4];
    const float* Wd1 = (const float*)d_in[5];
    const float* bd1 = (const float*)d_in[6];
    const float* Wi2 = (const float*)d_in[7];
    const float* Wh2 = (const float*)d_in[8];
    const float* b2  = (const float*)d_in[9];
    const float* Wd2 = (const float*)d_in[10];
    const float* bd2 = (const float*)d_in[11];
    const float* Wi3 = (const float*)d_in[12];
    const float* Wh3 = (const float*)d_in[13];
    const float* b3  = (const float*)d_in[14];
    const float* Wd3 = (const float*)d_in[15];
    const float* bd3 = (const float*)d_in[16];
    const float* Wd4 = (const float*)d_in[17];
    const float* bd4 = (const float*)d_in[18];

    const int BT = B_BATCH * T_SEQ;   // 204800

    // Deterministic bump allocation in workspace (1KB aligned).
    char* ws = (char*)d_ws;
    size_t off = 0;
    auto alloc = [&](size_t bytes) -> char* {
        char* p = ws + off;
        off += (bytes + 1023) & ~(size_t)1023;
        return p;
    };
    // Combined [Wi;Wh] packed fragment buffers (contiguous K-tile ranges).
    __bf16* W1c  = (__bf16*)alloc((size_t)(2 + 4) * 32 * 512 * 2);  // K=64+128, N=512
    __bf16* W2c  = (__bf16*)alloc((size_t)(8 + 2) * 16 * 512 * 2);  // K=256+64, N=256
    __bf16* W3c  = (__bf16*)alloc((size_t)(4 + 1) *  8 * 512 * 2);  // K=128+32, N=128
    __bf16* Wd1p = (__bf16*)alloc((size_t)4 * 16 * 512 * 2);        // K=128 N=256
    __bf16* Wd2p = (__bf16*)alloc((size_t)2 *  8 * 512 * 2);        // K=64  N=128
    __bf16* xe   = (__bf16*)alloc((size_t)BT *  64 * 2);
    __bf16* h1   = (__bf16*)alloc((size_t)BT * 128 * 2);
    __bf16* a1   = (__bf16*)alloc((size_t)BT * 256 * 2);
    __bf16* h2   = (__bf16*)alloc((size_t)BT *  64 * 2);
    __bf16* a2   = (__bf16*)alloc((size_t)BT * 128 * 2);
    __bf16* h3l  = (__bf16*)alloc((size_t)B_BATCH * 32 * 2);

    auto packLaunch = [&](const float* src, __bf16* dst, int srcK, int K, int N) {
        int total = (K / 32) * (N / 16) * 512;
        pack_w<<<(total + 255) / 256, 256, 0, stream>>>(src, dst, srcK, K, N);
    };
    // LSTM1: Wi (k-tiles 0..1, embed 50 padded to 64) then Wh (k-tiles 2..5).
    packLaunch(Wi1, W1c,                       50,  64, 512);
    packLaunch(Wh1, W1c + (size_t)2 * 32 * 512, 128, 128, 512);
    // LSTM2: Wi (k-tiles 0..7) then Wh (k-tiles 8..9).
    packLaunch(Wi2, W2c,                       256, 256, 256);
    packLaunch(Wh2, W2c + (size_t)8 * 16 * 512, 64,  64, 256);
    // LSTM3: Wi (k-tiles 0..3) then Wh (k-tile 4).
    packLaunch(Wi3, W3c,                       128, 128, 128);
    packLaunch(Wh3, W3c + (size_t)4 * 8 * 512,  32,  32, 128);
    // Dense weights.
    packLaunch(Wd1, Wd1p, 128, 128, 256);
    packLaunch(Wd2, Wd2p,  64,  64, 128);

    embed_gather<<<(BT * 64 + 255) / 256, 256, 0, stream>>>(x_batch, embed, xe, BT);

    // Dynamic LDS: weights + WAVES per-wave A-buffers (fits in 320KB/WGP).
    const size_t shm1 = ((size_t)6 * 32 * 512 + (size_t)4 * 16 * (64 + 128 + 8)) * 2;  // 222208
    const size_t shm2 = ((size_t)10 * 16 * 512 + (size_t)4 * 16 * (256 + 64 + 8)) * 2; // 205824
    const size_t shm3 = ((size_t)5 * 8 * 512 + (size_t)4 * 16 * (128 + 32 + 8)) * 2;   // 62464

    lstm_layer< 64, 128, 4><<<B_BATCH / 64, 128, shm1, stream>>>(xe, W1c, b1, h1, T_SEQ, 1);
    dense_relu<128, 256, 4><<<BT / 64, 128, 0, stream>>>(h1, Wd1p, bd1, a1, BT);
    lstm_layer<256,  64, 4><<<B_BATCH / 64, 128, shm2, stream>>>(a1, W2c, b2, h2, T_SEQ, 1);
    dense_relu< 64, 128, 4><<<BT / 64, 128, 0, stream>>>(h2, Wd2p, bd2, a2, BT);
    lstm_layer<128,  32, 4><<<B_BATCH / 64, 128, shm3, stream>>>(a2, W3c, b3, h3l, T_SEQ, 0);

    head_kernel<<<(B_BATCH + 127) / 128, 128, 0, stream>>>(h3l, Wd3, bd3, Wd4, bd4,
                                                           (float*)d_out, B_BATCH);
}
